// ChebNetwork_64707977281780
// MI455X (gfx1250) — matrix operand order repaired and checked by reading, not compiled, not run on hardware
//
#include <hip/hip_runtime.h>
#include <hip/hip_bf16.h>
#include <cstdint>

// ---------- types ----------
typedef __attribute__((ext_vector_type(16))) __bf16 v16bf;
typedef __attribute__((ext_vector_type(8)))  __bf16 v8bf;
typedef __attribute__((ext_vector_type(8)))  float  v8f;
typedef __attribute__((ext_vector_type(4)))  unsigned int u32x4;

#define FEAT 128
#define CATK 384   // 3 * 128

__device__ __forceinline__ unsigned short f2bf(float f) {
    unsigned int u = __float_as_uint(f);
    unsigned int r = u + 0x7FFFu + ((u >> 16) & 1u);  // round-to-nearest-even
    return (unsigned short)(r >> 16);
}

__device__ __forceinline__ void atomAddF(float* p, float v) {
#if defined(__AMDGCN__)
    unsafeAtomicAdd(p, v);      // lowers to global_atomic_add_f32
#else
    atomicAdd(p, v);
#endif
}

// ---------- small utility kernels ----------
__global__ void zero_f32(float* __restrict__ p, int n) {
    for (int i = blockIdx.x * blockDim.x + threadIdx.x; i < n;
         i += gridDim.x * blockDim.x)
        p[i] = 0.0f;
}

__global__ void deg_kernel(const int* __restrict__ src,
                           const float* __restrict__ ew,
                           float* __restrict__ deg, int nE) {
    int e = blockIdx.x * blockDim.x + threadIdx.x;
    if (e < nE) atomAddF(&deg[src[e]], ew[e]);
}

__global__ void dinv_kernel(float* __restrict__ deg, int nN) {
    int i = blockIdx.x * blockDim.x + threadIdx.x;
    if (i < nN) {
        float d = deg[i];
        deg[i] = (d > 0.0f) ? rsqrtf(fmaxf(d, 1e-30f)) : 0.0f;
    }
}

__global__ void norm_kernel(const int* __restrict__ src,
                            const int* __restrict__ dst,
                            const float* __restrict__ ew,
                            const float* __restrict__ dinv,
                            float* __restrict__ nrm, int nE) {
    int e = blockIdx.x * blockDim.x + threadIdx.x;
    if (e < nE) nrm[e] = -dinv[src[e]] * ew[e] * dinv[dst[e]];
}

// ---------- SpMM: out[dst] += norm * x[src], one wave per edge ----------
__global__ void spmm_kernel(const float* __restrict__ x,
                            const int* __restrict__ src,
                            const int* __restrict__ dst,
                            const float* __restrict__ nrm,
                            float* __restrict__ out, int nE) {
    int t = blockIdx.x * blockDim.x + threadIdx.x;
    int e = t >> 5;
    if (e >= nE) return;
    int lane = t & 31;
    float w = nrm[e];
    const float4* xr = (const float4*)(x + (size_t)src[e] * FEAT);
    float4 v = xr[lane];                              // coalesced 512B per edge
    float* o = out + (size_t)dst[e] * FEAT + lane * 4;
    atomAddF(o + 0, w * v.x);
    atomAddF(o + 1, w * v.y);
    atomAddF(o + 2, w * v.z);
    atomAddF(o + 3, w * v.w);
}

// tx2 = 2*tx2 - xin   (Chebyshev recurrence finalize)
__global__ void cheb_finalize(float* __restrict__ tx2,
                              const float* __restrict__ xin, int n) {
    int i = blockIdx.x * blockDim.x + threadIdx.x;
    if (i < n) tx2[i] = 2.0f * tx2[i] - xin[i];
}

// pack [xin | tx1 | tx2] rows into N x 384 bf16 matrix
__global__ void cat_bf16_kernel(const float* __restrict__ xin,
                                const float* __restrict__ tx1,
                                const float* __restrict__ tx2,
                                unsigned short* __restrict__ cat, int nN) {
    int idx = blockIdx.x * blockDim.x + threadIdx.x;
    int total = nN * CATK;
    if (idx >= total) return;
    int row = idx / CATK;
    int c   = idx - row * CATK;
    float v;
    if (c < FEAT)            v = xin[(size_t)row * FEAT + c];
    else if (c < 2 * FEAT)   v = tx1[(size_t)row * FEAT + c - FEAT];
    else                     v = tx2[(size_t)row * FEAT + c - 2 * FEAT];
    cat[idx] = f2bf(v);
}

// f32 -> bf16 elementwise (weights: flat (3,128,O) == stacked (384,O))
__global__ void w_bf16_kernel(const float* __restrict__ w,
                              unsigned short* __restrict__ wb, int n) {
    int i = blockIdx.x * blockDim.x + threadIdx.x;
    if (i < n) wb[i] = f2bf(w[i]);
}

// ---------- fused GEMM + bias + sigmoid using v_wmma_f32_16x16x32_bf16 ----------
// cat: N x 384 bf16 (row major), w: 384 x O bf16 (row major), out: N x O f32
// One wave per 16x16 output tile; K loop = 12 WMMAs.
__global__ void gemm_wmma_kernel(const unsigned short* __restrict__ cat,
                                 const unsigned short* __restrict__ w,
                                 const float* __restrict__ bias,
                                 float* __restrict__ out,
                                 int nRows, int O) {
    const int lane   = threadIdx.x & 31;
    const int wave   = threadIdx.x >> 5;
    const int tilesN = O >> 4;
    const int tiles  = (nRows >> 4) * tilesN;
    const int tile   = blockIdx.x * (blockDim.x >> 5) + wave;  // wave-uniform
    if (tile >= tiles) return;                                 // whole wave exits
    const int tileM = tile / tilesN;
    const int tileN = tile - tileM * tilesN;

    const int m = lane & 15;       // A row within tile / D column
    const int g = lane >> 4;       // lane group selects K-halves (A) / K-rows (B)

    // A: row-major node features; per ISA 16-bit A 16x32 layout:
    //   group 0 halves0-7 = K 0..7, halves8-15 = K 16..23 ; group 1 shifted by 8
    const unsigned short* arow =
        cat + (size_t)(tileM * 16 + m) * CATK + g * 8;
    // B: lane holds N column (lane&15), 16 halves = K in same order as A
    const int kl    = (lane & 15) + (g << 4);
    const int nBase = tileN * 16;

    v8f c = {};
#pragma unroll
    for (int kb = 0; kb < CATK; kb += 32) {
        u32x4 a0 = *reinterpret_cast<const u32x4*>(arow + kb);
        u32x4 a1 = *reinterpret_cast<const u32x4*>(arow + kb + 16);
        const unsigned short* bcol = w + (size_t)(kb + g * 8) * O + nBase + (lane & 15);
        u32x4 b0, b1;
#pragma unroll
        for (int i = 0; i < 4; ++i) {
            b0[i] = (unsigned int)bcol[(size_t)(2 * i) * O] |
                    ((unsigned int)bcol[(size_t)(2 * i + 1) * O] << 16);
            b1[i] = (unsigned int)bcol[(size_t)(16 + 2 * i) * O] |
                    ((unsigned int)bcol[(size_t)(17 + 2 * i) * O] << 16);
        }

        v8bf alo = __builtin_bit_cast(v8bf, a0);
        v8bf ahi = __builtin_bit_cast(v8bf, a1);
        v8bf blo = __builtin_bit_cast(v8bf, b0);
        v8bf bhi = __builtin_bit_cast(v8bf, b1);
        v16bf av = __builtin_shufflevector(alo, ahi, 0,1,2,3,4,5,6,7,
                                                     8,9,10,11,12,13,14,15);
        v16bf bv = __builtin_shufflevector(blo, bhi, 0,1,2,3,4,5,6,7,
                                                     8,9,10,11,12,13,14,15);
        c = __builtin_amdgcn_wmma_f32_16x16x32_bf16(
                false, av, false, bv, (short)0, c, false, false);
    }

    // D layout: lane group g, VGPR r -> row M = r + 8*g, col N = lane&15
    const int col  = nBase + (lane & 15);
    const float bc = bias[col];
    const int row0 = tileM * 16 + (g << 3);
#pragma unroll
    for (int r = 0; r < 8; ++r) {
        float v = c[r] + bc;
        out[(size_t)(row0 + r) * O + col] = 1.0f / (1.0f + __expf(-v));
    }
}

// ---------- host orchestration ----------
extern "C" void kernel_launch(void* const* d_in, const int* in_sizes, int n_in,
                              void* d_out, int out_size, void* d_ws, size_t ws_size,
                              hipStream_t stream) {
    const float* x  = (const float*)d_in[0];
    const int*   ei = (const int*)d_in[1];
    const float* ew = (const float*)d_in[2];
    const int nN = in_sizes[0] / FEAT;     // 50000
    const int nE = in_sizes[2];            // 1600000
    const int* src = ei;
    const int* dst = ei + nE;

    // workspace carve-up (256B aligned)
    char* wsp = (char*)d_ws;
    size_t off = 0;
    auto alloc = [&](size_t bytes) -> void* {
        void* p = wsp + off;
        off = (off + bytes + 255) & ~(size_t)255;
        return p;
    };
    float*          d_norm = (float*)alloc((size_t)nE * 4);
    float*          d_dinv = (float*)alloc((size_t)nN * 4);   // deg then dinv
    float*          d_tx1  = (float*)alloc((size_t)nN * FEAT * 4);
    float*          d_tx2  = (float*)alloc((size_t)nN * FEAT * 4);
    float*          d_h    = (float*)alloc((size_t)nN * FEAT * 4);
    unsigned short* d_cat  = (unsigned short*)alloc((size_t)nN * CATK * 2);
    unsigned short* d_wbf  = (unsigned short*)alloc((size_t)CATK * FEAT * 2);

    const int B = 256;
    const int nF = nN * FEAT;

    // 1. normalized Laplacian edge weights
    zero_f32<<<1024, B, 0, stream>>>(d_dinv, nN);
    deg_kernel<<<(nE + B - 1) / B, B, 0, stream>>>(src, ew, d_dinv, nE);
    dinv_kernel<<<(nN + B - 1) / B, B, 0, stream>>>(d_dinv, nN);
    norm_kernel<<<(nE + B - 1) / B, B, 0, stream>>>(src, dst, ew, d_dinv, d_norm, nE);

    const int spmmBlocks = (int)(((size_t)nE * 32 + B - 1) / B);

    for (int layer = 0; layer < 3; ++layer) {
        const float* xin  = (layer == 0) ? x : d_h;
        const int    O    = (layer < 2) ? 128 : 64;
        const float* W    = (const float*)d_in[3 + 2 * layer];
        const float* bias = (const float*)d_in[4 + 2 * layer];
        float*       outp = (layer < 2) ? d_h : (float*)d_out;

        // tx1 = L_hat @ xin
        zero_f32<<<1024, B, 0, stream>>>(d_tx1, nF);
        spmm_kernel<<<spmmBlocks, B, 0, stream>>>(xin, src, dst, d_norm, d_tx1, nE);
        // tx2 = 2 * L_hat @ tx1 - xin
        zero_f32<<<1024, B, 0, stream>>>(d_tx2, nF);
        spmm_kernel<<<spmmBlocks, B, 0, stream>>>(d_tx1, src, dst, d_norm, d_tx2, nE);
        cheb_finalize<<<(nF + B - 1) / B, B, 0, stream>>>(d_tx2, xin, nF);

        // pack bf16 operands
        w_bf16_kernel<<<(CATK * O + B - 1) / B, B, 0, stream>>>(W, d_wbf, CATK * O);
        cat_bf16_kernel<<<((size_t)nN * CATK + B - 1) / B, B, 0, stream>>>(
            xin, d_tx1, d_tx2, d_cat, nN);

        // fused 3-term GEMM + bias + sigmoid (reads cat only, so it may
        // safely overwrite d_h in place)
        const int tiles  = (nN / 16) * (O / 16);
        const int blocks = (tiles + 7) / 8;   // 8 waves (16x16 tiles) per block
        gemm_wmma_kernel<<<blocks, 256, 0, stream>>>(d_cat, d_wbf, bias, outp, nN, O);
    }
}
